// TransformerBlock_3375844295198
// MI455X (gfx1250) — compile-verified, matrix operand order checked by reading, not run on hardware
//
#include <hip/hip_runtime.h>
#include <math.h>

typedef __attribute__((ext_vector_type(16))) __bf16 bf16x16;
typedef __attribute__((ext_vector_type(8)))  __bf16 bf16x8;
typedef __attribute__((ext_vector_type(8)))  float  floatx8;

#define BN_EPS 1e-5f

// ---------------------------------------------------------------------------
// small utility: zero a float buffer (stats region)
// ---------------------------------------------------------------------------
__global__ void zero_f_kernel(float* p, int n) {
    int i = blockIdx.x * blockDim.x + threadIdx.x;
    if (i < n) p[i] = 0.0f;
}

// ---------------------------------------------------------------------------
// convert f32 -> bf16 (weights), grid-stride
// ---------------------------------------------------------------------------
__global__ void f32_to_bf16_kernel(const float* __restrict__ src,
                                   __bf16* __restrict__ dst, int n) {
    for (int i = blockIdx.x * blockDim.x + threadIdx.x; i < n;
         i += gridDim.x * blockDim.x)
        dst[i] = (__bf16)src[i];
}

// ---------------------------------------------------------------------------
// h = relu(bn1(h1)) materialized once in bf16 (shared A of 3 projections)
// ---------------------------------------------------------------------------
__global__ void bnrelu_bf16_kernel(const float* __restrict__ h1,
                                   const float* __restrict__ bnSum,
                                   const float* __restrict__ bnSq,
                                   const float* __restrict__ bnG,
                                   const float* __restrict__ bnB,
                                   float invCnt, __bf16* __restrict__ hbf,
                                   int total) {
    for (int i = blockIdx.x * blockDim.x + threadIdx.x; i < total;
         i += gridDim.x * blockDim.x) {
        int c = i & 127;
        float m = bnSum[c] * invCnt;
        float v = bnSq[c] * invCnt - m * m;
        float s = bnG[c] * rsqrtf(v + BN_EPS);
        float t = bnB[c] - m * s;
        hbf[i] = (__bf16)fmaxf(fmaf(h1[i], s, t), 0.0f);
    }
}

// ---------------------------------------------------------------------------
// fragment load from packed bf16 row: elements [k0,k0+8) and [k0+16,k0+24)
// (CDNA5 16-bit A/B 16x32 per-lane layout; 16B-aligned vector loads)
// ---------------------------------------------------------------------------
__device__ __forceinline__ bf16x16 load_frag_bf16(const __bf16* base, int k0) {
    bf16x8 lo = *(const bf16x8*)(base + k0);
    bf16x8 hi = *(const bf16x8*)(base + k0 + 16);
    bf16x16 r;
    #pragma unroll
    for (int i = 0; i < 8; ++i) { r[i] = lo[i]; r[8 + i] = hi[i]; }
    return r;
}

// epilogue: store one 16x16 C tile; uniform fast path when tile fully in-range
__device__ __forceinline__ void store_tile(const floatx8& acc, int rowBase,
                                           int colBase, int r, int hf, float bv,
                                           float* __restrict__ out, int M) {
    if (rowBase + 16 <= M) {
        #pragma unroll
        for (int j = 0; j < 8; ++j) {
            int orow = rowBase + j + 8 * hf;
            out[(size_t)orow * 128 + colBase + r] = acc[j] + bv;
        }
    } else {
        #pragma unroll
        for (int j = 0; j < 8; ++j) {
            int orow = rowBase + j + 8 * hf;
            if (orow < M) out[(size_t)orow * 128 + colBase + r] = acc[j] + bv;
        }
    }
}

// ---------------------------------------------------------------------------
// WMMA GEMM, A in f32 (optionally BN+relu fused on load), W pre-converted bf16.
// block = 256 threads = 8 waves; wave w owns cols [16w,16w+16); each wave does
// TWO 16-row tiles (rows [32b,32b+32)); 4 K-steps of wmma_f32_16x16x32_bf16.
// ---------------------------------------------------------------------------
template <bool DO_BN>
__global__ void gemm_f32A_wmma_kernel(const float* __restrict__ A,
                                      const __bf16* __restrict__ Wbf,
                                      const float* __restrict__ bias,
                                      const float* __restrict__ bnSum,
                                      const float* __restrict__ bnSq,
                                      const float* __restrict__ bnG,
                                      const float* __restrict__ bnB,
                                      float invCnt,
                                      float* __restrict__ out, int M) {
    __shared__ float sSc[128];
    __shared__ float sSh[128];
    int tid = threadIdx.x;
    if (DO_BN) {
        if (tid < 128) {
            float mean = bnSum[tid] * invCnt;
            float var  = bnSq[tid] * invCnt - mean * mean;
            float s = bnG[tid] * rsqrtf(var + BN_EPS);
            sSc[tid] = s;
            sSh[tid] = bnB[tid] - mean * s;
        }
        __syncthreads();
    }
    int wave = tid >> 5;
    int lane = tid & 31;
    int r  = lane & 15;
    int hf = lane >> 4;
    int rowBase0 = blockIdx.x * 32;
    int rowBase1 = rowBase0 + 16;
    int colBase = wave * 16;

    int row0 = rowBase0 + r;
    int row1 = rowBase1 + r;
    const float*  Arow0 = A + (size_t)((row0 < M) ? row0 : (M - 1)) * 128;
    const float*  Arow1 = A + (size_t)((row1 < M) ? row1 : (M - 1)) * 128;
    const __bf16* Wrow  = Wbf + (size_t)(colBase + r) * 128;

    floatx8 acc0 = {};
    floatx8 acc1 = {};
    for (int kk = 0; kk < 4; ++kk) {
        int k0 = kk * 32 + hf * 8;
        bf16x16 bfr = load_frag_bf16(Wrow, k0);
        bf16x16 af0, af1;
        #pragma unroll
        for (int i = 0; i < 16; ++i) {
            int k = k0 + ((i < 8) ? i : (8 + i));   // k0+i / k0+16+(i-8)
            float a = Arow0[k];
            float c = Arow1[k];
            if (DO_BN) {
                a = fmaxf(fmaf(a, sSc[k], sSh[k]), 0.0f);
                c = fmaxf(fmaf(c, sSc[k], sSh[k]), 0.0f);
            }
            af0[i] = (__bf16)a;
            af1[i] = (__bf16)c;
        }
        acc0 = __builtin_amdgcn_wmma_f32_16x16x32_bf16(
            false, af0, false, bfr, (short)0, acc0, false, false);
        acc1 = __builtin_amdgcn_wmma_f32_16x16x32_bf16(
            false, af1, false, bfr, (short)0, acc1, false, false);
    }
    float bv = bias ? bias[colBase + r] : 0.0f;
    store_tile(acc0, rowBase0, colBase, r, hf, bv, out, M);
    store_tile(acc1, rowBase1, colBase, r, hf, bv, out, M);
}

// ---------------------------------------------------------------------------
// WMMA GEMM, A and W both pre-converted bf16 (3 projection GEMMs)
// ---------------------------------------------------------------------------
__global__ void gemm_bf16A_wmma_kernel(const __bf16* __restrict__ Abf,
                                       const __bf16* __restrict__ Wbf,
                                       const float* __restrict__ bias,
                                       float* __restrict__ out, int M) {
    int tid = threadIdx.x;
    int wave = tid >> 5;
    int lane = tid & 31;
    int r  = lane & 15;
    int hf = lane >> 4;
    int rowBase0 = blockIdx.x * 32;
    int rowBase1 = rowBase0 + 16;
    int colBase = wave * 16;

    int row0 = rowBase0 + r;
    int row1 = rowBase1 + r;
    const __bf16* Arow0 = Abf + (size_t)((row0 < M) ? row0 : (M - 1)) * 128;
    const __bf16* Arow1 = Abf + (size_t)((row1 < M) ? row1 : (M - 1)) * 128;
    const __bf16* Wrow  = Wbf + (size_t)(colBase + r) * 128;

    floatx8 acc0 = {};
    floatx8 acc1 = {};
    #pragma unroll
    for (int kk = 0; kk < 4; ++kk) {
        int k0 = kk * 32 + hf * 8;
        bf16x16 af0 = load_frag_bf16(Arow0, k0);
        bf16x16 af1 = load_frag_bf16(Arow1, k0);
        bf16x16 bfr = load_frag_bf16(Wrow, k0);
        acc0 = __builtin_amdgcn_wmma_f32_16x16x32_bf16(
            false, af0, false, bfr, (short)0, acc0, false, false);
        acc1 = __builtin_amdgcn_wmma_f32_16x16x32_bf16(
            false, af1, false, bfr, (short)0, acc1, false, false);
    }
    float bv = bias ? bias[colBase + r] : 0.0f;
    store_tile(acc0, rowBase0, colBase, r, hf, bv, out, M);
    store_tile(acc1, rowBase1, colBase, r, hf, bv, out, M);
}

// ---------------------------------------------------------------------------
// per-column sum / sum-of-squares over [M,128]
// ---------------------------------------------------------------------------
__global__ void colstats128_kernel(const float* __restrict__ X, int M,
                                   float* __restrict__ sum, float* __restrict__ sq) {
    int c = threadIdx.x;   // 128 threads
    float s = 0.0f, q = 0.0f;
    for (int r = blockIdx.x; r < M; r += gridDim.x) {
        float v = X[(size_t)r * 128 + c];
        s += v;
        q += v * v;
    }
    atomicAdd(&sum[c], s);
    atomicAdd(&sq[c], q);
}

// ---------------------------------------------------------------------------
// bnp helpers: per-thread preload of pos-MLP weights + bnp scale/shift
// ---------------------------------------------------------------------------
__device__ __forceinline__ void load_bnp(const float* __restrict__ Wp1,
                                         const float* __restrict__ bp1,
                                         const float* __restrict__ Psum,
                                         const float* __restrict__ Psq,
                                         const float* __restrict__ gP,
                                         const float* __restrict__ bP,
                                         float invE,
                                         float* w1, float* b1v, float* sP, float* tP) {
    #pragma unroll
    for (int i = 0; i < 9; ++i) w1[i] = Wp1[i];
    #pragma unroll
    for (int j = 0; j < 3; ++j) {
        b1v[j] = bp1[j];
        float m = Psum[j] * invE;
        float v = Psq[j] * invE - m * m;
        float s = gP[j] * rsqrtf(v + BN_EPS);
        sP[j] = s;
        tP[j] = bP[j] - m * s;
    }
}

__device__ __forceinline__ float compute_delta(float rx, float ry, float rz,
                                               const float* w1, const float* b1v,
                                               const float* sP, const float* tP,
                                               const float* wp2r, float bp2c) {
    float d = bp2c;
    #pragma unroll
    for (int j = 0; j < 3; ++j) {
        float u = b1v[j] + w1[j * 3] * rx + w1[j * 3 + 1] * ry + w1[j * 3 + 2] * rz;
        float t3 = fmaxf(fmaf(u, sP[j], tP[j]), 0.0f);
        d = fmaf(wp2r[j], t3, d);
    }
    return d;
}

// ---------------------------------------------------------------------------
// edge pass 1: stats of u = rel @ Wp1^T + bp1 over all E edges (3 channels)
// ---------------------------------------------------------------------------
__global__ void edge_bnp_stats_kernel(const int* __restrict__ src,
                                      const int* __restrict__ dst,
                                      const float* __restrict__ pos,
                                      const float* __restrict__ Wp1,
                                      const float* __restrict__ bp1,
                                      int E, float* __restrict__ sumP,
                                      float* __restrict__ sqP) {
    __shared__ float red[6];
    if (threadIdx.x < 6) red[threadIdx.x] = 0.0f;
    __syncthreads();
    float w1[9], b1v[3];
    #pragma unroll
    for (int i = 0; i < 9; ++i) w1[i] = Wp1[i];
    #pragma unroll
    for (int j = 0; j < 3; ++j) b1v[j] = bp1[j];
    float ls[3] = {0, 0, 0}, lq[3] = {0, 0, 0};
    for (int e = blockIdx.x * blockDim.x + threadIdx.x; e < E;
         e += gridDim.x * blockDim.x) {
        int si = src[e], di = dst[e];
        float rx = pos[di * 3 + 0] - pos[si * 3 + 0];
        float ry = pos[di * 3 + 1] - pos[si * 3 + 1];
        float rz = pos[di * 3 + 2] - pos[si * 3 + 2];
        #pragma unroll
        for (int j = 0; j < 3; ++j) {
            float u = b1v[j] + w1[j * 3] * rx + w1[j * 3 + 1] * ry + w1[j * 3 + 2] * rz;
            ls[j] += u;
            lq[j] += u * u;
        }
    }
    #pragma unroll
    for (int j = 0; j < 3; ++j) {
        atomicAdd(&red[j], ls[j]);
        atomicAdd(&red[3 + j], lq[j]);
    }
    __syncthreads();
    if (threadIdx.x < 3) {
        atomicAdd(&sumP[threadIdx.x], red[threadIdx.x]);
        atomicAdd(&sqP[threadIdx.x], red[3 + threadIdx.x]);
    }
}

// ---------------------------------------------------------------------------
// edge pass 2: bna1 stats over alpha = a_dst[n] - a_src[src] + delta
// ---------------------------------------------------------------------------
#define NPB2 16
__global__ void edge_bna1_stats_kernel(const int* __restrict__ src,
                                       const float* __restrict__ pos,
                                       const float* __restrict__ aSrc,
                                       const float* __restrict__ aDst,
                                       const float* __restrict__ Wp1,
                                       const float* __restrict__ bp1,
                                       const float* __restrict__ Wp2,
                                       const float* __restrict__ bp2,
                                       const float* __restrict__ Psum,
                                       const float* __restrict__ Psq,
                                       const float* __restrict__ gP,
                                       const float* __restrict__ bP,
                                       float invE, int N, int Kdeg,
                                       float* __restrict__ sumA,
                                       float* __restrict__ sqA) {
    int c = threadIdx.x;
    float w1[9], b1v[3], sP[3], tP[3];
    load_bnp(Wp1, bp1, Psum, Psq, gP, bP, invE, w1, b1v, sP, tP);
    float wp2r[3] = {Wp2[c * 3], Wp2[c * 3 + 1], Wp2[c * 3 + 2]};
    float bp2c = bp2[c];
    float s = 0.0f, q = 0.0f;
    int n0 = blockIdx.x * NPB2;
    for (int ni = 0; ni < NPB2; ++ni) {
        int n = n0 + ni;
        if (n >= N) break;
        float dC = aDst[(size_t)n * 128 + c];
        float pdx = pos[n * 3], pdy = pos[n * 3 + 1], pdz = pos[n * 3 + 2];
        for (int el = 0; el <= Kdeg; ++el) {
            int e = (el < Kdeg) ? (n * Kdeg + el) : (N * Kdeg + n);
            int si = src[e];
            __builtin_prefetch(&aSrc[(size_t)si * 128 + c], 0, 1);
            float rx = pdx - pos[si * 3], ry = pdy - pos[si * 3 + 1],
                  rz = pdz - pos[si * 3 + 2];
            float delta = compute_delta(rx, ry, rz, w1, b1v, sP, tP, wp2r, bp2c);
            float alpha = dC - aSrc[(size_t)si * 128 + c] + delta;
            s += alpha;
            q += alpha * alpha;
        }
    }
    atomicAdd(&sumA[c], s);
    atomicAdd(&sqA[c], q);
}

// ---------------------------------------------------------------------------
// edge pass 3: t2[e,16] = relu(bna1(alpha)) @ Wa1^T + ba1 ; bna2 stats
// ---------------------------------------------------------------------------
#define NPB3 16
__global__ void edge_t2_kernel(const int* __restrict__ src,
                               const float* __restrict__ pos,
                               const float* __restrict__ aSrc,
                               const float* __restrict__ aDst,
                               const float* __restrict__ Wp1,
                               const float* __restrict__ bp1,
                               const float* __restrict__ Wp2,
                               const float* __restrict__ bp2,
                               const float* __restrict__ Psum,
                               const float* __restrict__ Psq,
                               const float* __restrict__ gP,
                               const float* __restrict__ bP,
                               const float* __restrict__ A1sum,
                               const float* __restrict__ A1sq,
                               const float* __restrict__ gA1,
                               const float* __restrict__ bA1,
                               const float* __restrict__ Wa1,
                               const float* __restrict__ ba1,
                               float invE, int N, int Kdeg,
                               float* __restrict__ t2,
                               float* __restrict__ sumA2,
                               float* __restrict__ sqA2) {
    __shared__ float lds_a1[17 * 132];
    __shared__ float ldsWa1[16 * 132];
    __shared__ float ldsred[32];
    int t = threadIdx.x;
    int c = t;
    float w1[9], b1v[3], sP[3], tP[3];
    load_bnp(Wp1, bp1, Psum, Psq, gP, bP, invE, w1, b1v, sP, tP);
    float wp2r[3] = {Wp2[c * 3], Wp2[c * 3 + 1], Wp2[c * 3 + 2]};
    float bp2c = bp2[c];
    float mA = A1sum[c] * invE;
    float vA = A1sq[c] * invE - mA * mA;
    float sA = gA1[c] * rsqrtf(vA + BN_EPS);
    float tA = bA1[c] - mA * sA;
    for (int idx = t; idx < 16 * 128; idx += 128)
        ldsWa1[(idx >> 7) * 132 + (idx & 127)] = Wa1[idx];
    __syncthreads();
    int j = t & 15, eg = t >> 4;
    float ba1j = ba1[j];
    float sj = 0.0f, qj = 0.0f;
    int n0 = blockIdx.x * NPB3;
    for (int ni = 0; ni < NPB3; ++ni) {
        int n = n0 + ni;
        if (n >= N) break;
        float dC = aDst[(size_t)n * 128 + c];
        float pdx = pos[n * 3], pdy = pos[n * 3 + 1], pdz = pos[n * 3 + 2];
        for (int el = 0; el <= Kdeg; ++el) {
            int e = (el < Kdeg) ? (n * Kdeg + el) : (N * Kdeg + n);
            int si = src[e];
            __builtin_prefetch(&aSrc[(size_t)si * 128 + c], 0, 1);
            float rx = pdx - pos[si * 3], ry = pdy - pos[si * 3 + 1],
                  rz = pdz - pos[si * 3 + 2];
            float delta = compute_delta(rx, ry, rz, w1, b1v, sP, tP, wp2r, bp2c);
            float alpha = dC - aSrc[(size_t)si * 128 + c] + delta;
            lds_a1[el * 132 + c] = fmaxf(fmaf(alpha, sA, tA), 0.0f);
        }
        __syncthreads();
        #pragma unroll
        for (int rep = 0; rep < 3; ++rep) {
            int e = eg + rep * 8;
            if (e <= Kdeg) {
                float acc = ba1j;
                #pragma unroll 8
                for (int cc = 0; cc < 128; ++cc)
                    acc = fmaf(lds_a1[e * 132 + cc], ldsWa1[j * 132 + cc], acc);
                int ge = (e < Kdeg) ? (n * Kdeg + e) : (N * Kdeg + n);
                t2[(size_t)ge * 16 + j] = acc;
                sj += acc;
                qj += acc * acc;
            }
        }
        __syncthreads();
    }
    if (t < 32) ldsred[t] = 0.0f;
    __syncthreads();
    atomicAdd(&ldsred[j], sj);
    atomicAdd(&ldsred[16 + j], qj);
    __syncthreads();
    if (t < 16) {
        atomicAdd(&sumA2[t], ldsred[t]);
        atomicAdd(&sqA2[t], ldsred[16 + t]);
    }
}

// ---------------------------------------------------------------------------
// edge pass 4: attention logits, per-node softmax, weighted aggregation
// ---------------------------------------------------------------------------
#define NPB4 4
__global__ void edge_attn_agg_kernel(const int* __restrict__ src,
                                     const float* __restrict__ pos,
                                     const float* __restrict__ xLin,
                                     const float* __restrict__ t2,
                                     const float* __restrict__ Wp1,
                                     const float* __restrict__ bp1,
                                     const float* __restrict__ Wp2,
                                     const float* __restrict__ bp2,
                                     const float* __restrict__ Psum,
                                     const float* __restrict__ Psq,
                                     const float* __restrict__ gP,
                                     const float* __restrict__ bP,
                                     const float* __restrict__ A2sum,
                                     const float* __restrict__ A2sq,
                                     const float* __restrict__ gA2,
                                     const float* __restrict__ bA2,
                                     const float* __restrict__ Wa2,
                                     const float* __restrict__ ba2,
                                     float invE, int N, int Kdeg,
                                     float* __restrict__ agg) {
    __shared__ float latt[17 * 20];
    int t = threadIdx.x;
    int c = t;
    int j = t & 15, eg = t >> 4, q = c & 15;
    float w1[9], b1v[3], sP[3], tP[3];
    load_bnp(Wp1, bp1, Psum, Psq, gP, bP, invE, w1, b1v, sP, tP);
    float wp2r[3] = {Wp2[c * 3], Wp2[c * 3 + 1], Wp2[c * 3 + 2]};
    float bp2c = bp2[c];
    float sA2[16], tA2[16], wa2r[16];
    #pragma unroll
    for (int k = 0; k < 16; ++k) {
        float m = A2sum[k] * invE;
        float v = A2sq[k] * invE - m * m;
        float s = gA2[k] * rsqrtf(v + BN_EPS);
        sA2[k] = s;
        tA2[k] = bA2[k] - m * s;
        wa2r[k] = Wa2[j * 16 + k];
    }
    float ba2j = ba2[j];
    float invDeg = 1.0f / (float)(Kdeg + 1);
    int n0 = blockIdx.x * NPB4;
    for (int ni = 0; ni < NPB4; ++ni) {
        int n = n0 + ni;
        if (n >= N) break;
        #pragma unroll
        for (int rep = 0; rep < 3; ++rep) {
            int e = eg + rep * 8;
            if (e <= Kdeg) {
                int ge = (e < Kdeg) ? (n * Kdeg + e) : (N * Kdeg + n);
                const float* t2r = t2 + (size_t)ge * 16;
                float acc = ba2j;
                #pragma unroll
                for (int k = 0; k < 16; ++k) {
                    float a2 = fmaxf(fmaf(t2r[k], sA2[k], tA2[k]), 0.0f);
                    acc = fmaf(a2, wa2r[k], acc);
                }
                latt[e * 20 + j] = acc;
            }
        }
        __syncthreads();
        if (t < 16) {
            float m = -3.0e38f;
            for (int e = 0; e <= Kdeg; ++e) m = fmaxf(m, latt[e * 20 + t]);
            float ssum = 0.0f;
            for (int e = 0; e <= Kdeg; ++e) {
                float ex = __expf(latt[e * 20 + t] - m);
                latt[e * 20 + t] = ex;
                ssum += ex;
            }
            float inv = 1.0f / (ssum + 1e-16f);
            for (int e = 0; e <= Kdeg; ++e) latt[e * 20 + t] *= inv;
        }
        __syncthreads();
        float pdx = pos[n * 3], pdy = pos[n * 3 + 1], pdz = pos[n * 3 + 2];
        float acc = 0.0f;
        for (int el = 0; el <= Kdeg; ++el) {
            int e = (el < Kdeg) ? (n * Kdeg + el) : (N * Kdeg + n);
            int si = src[e];
            __builtin_prefetch(&xLin[(size_t)si * 128 + c], 0, 1);
            float rx = pdx - pos[si * 3], ry = pdy - pos[si * 3 + 1],
                  rz = pdz - pos[si * 3 + 2];
            float delta = compute_delta(rx, ry, rz, w1, b1v, sP, tP, wp2r, bp2c);
            float xl = xLin[(size_t)si * 128 + c];
            acc = fmaf(latt[el * 20 + q], xl + delta, acc);
        }
        agg[(size_t)n * 128 + c] = acc * invDeg;
        __syncthreads();
    }
}

// ---------------------------------------------------------------------------
// final: out = relu(bn3(h3) + x_skip)
// ---------------------------------------------------------------------------
__global__ void final_bn_res_kernel(const float* __restrict__ h3,
                                    const float* __restrict__ x,
                                    const float* __restrict__ S3sum,
                                    const float* __restrict__ S3sq,
                                    const float* __restrict__ g3,
                                    const float* __restrict__ b3,
                                    float invN, float* __restrict__ out, int total) {
    for (int i = blockIdx.x * blockDim.x + threadIdx.x; i < total;
         i += gridDim.x * blockDim.x) {
        int c = i & 127;
        float m = S3sum[c] * invN;
        float v = S3sq[c] * invN - m * m;
        float s = g3[c] * rsqrtf(v + BN_EPS);
        float tt = b3[c] - m * s;
        out[i] = fmaxf(fmaf(h3[i], s, tt) + x[i], 0.0f);
    }
}

// ---------------------------------------------------------------------------
// host launch
// ---------------------------------------------------------------------------
extern "C" void kernel_launch(void* const* d_in, const int* in_sizes, int n_in,
                              void* d_out, int out_size, void* d_ws, size_t ws_size,
                              hipStream_t stream) {
    const float* x   = (const float*)d_in[0];
    const float* pos = (const float*)d_in[1];
    const int*   ei  = (const int*)d_in[2];
    int idx = 3;
    const float* W_in  = (const float*)d_in[idx++];
    const float* bn1_g = (const float*)d_in[idx++];
    const float* bn1_b = (const float*)d_in[idx++];
    const float* Wl    = (const float*)d_in[idx++];
    const float* bl    = (const float*)d_in[idx++];
    const float* Wsrc  = (const float*)d_in[idx++];
    const float* bsrc  = (const float*)d_in[idx++];
    const float* Wdst  = (const float*)d_in[idx++];
    const float* bdst  = (const float*)d_in[idx++];
    const float* Wp1   = (const float*)d_in[idx++];
    const float* bp1   = (const float*)d_in[idx++];
    const float* bnp_g = (const float*)d_in[idx++];
    const float* bnp_b = (const float*)d_in[idx++];
    const float* Wp2   = (const float*)d_in[idx++];
    const float* bp2   = (const float*)d_in[idx++];
    const float* bna1_g = (const float*)d_in[idx++];
    const float* bna1_b = (const float*)d_in[idx++];
    const float* Wa1   = (const float*)d_in[idx++];
    const float* ba1   = (const float*)d_in[idx++];
    const float* bna2_g = (const float*)d_in[idx++];
    const float* bna2_b = (const float*)d_in[idx++];
    const float* Wa2   = (const float*)d_in[idx++];
    const float* ba2   = (const float*)d_in[idx++];
    const float* bn2_g = (const float*)d_in[idx++];
    const float* bn2_b = (const float*)d_in[idx++];
    const float* W_out = (const float*)d_in[idx++];
    const float* bn3_g = (const float*)d_in[idx++];
    const float* bn3_b = (const float*)d_in[idx++];

    int N = in_sizes[0] / 128;
    int E = in_sizes[2] / 2;
    int Kdeg = (E - N) / N;   // 16
    const int* srcArr = ei;
    const int* dstArr = ei + E;

    float* ws = (float*)d_ws;
    // stats layout (floats)
    float* S1s = ws + 0;    float* S1q = ws + 128;
    float* Ps  = ws + 256;  float* Pq  = ws + 260;
    float* A1s = ws + 264;  float* A1q = ws + 392;
    float* A2s = ws + 520;  float* A2q = ws + 536;
    float* S2s = ws + 552;  float* S2q = ws + 680;
    float* S3s = ws + 808;  float* S3q = ws + 936;
    const int nStats = 1088;
    float* bufs = ws + nStats;
    size_t NC = (size_t)N * 128;
    float* h1    = bufs;
    float* xlin  = h1 + NC;
    float* asrcB = xlin + NC;
    float* adstB = asrcB + NC;
    float* agg   = adstB + NC;
    float* h3    = agg + NC;
    float* t2    = h3 + NC;                       // E*16 floats
    __bf16* hbf  = (__bf16*)(t2 + (size_t)E * 16); // N*128 bf16
    __bf16* wbf  = hbf + NC;                       // 5 * 128*128 bf16
    __bf16* wb_in  = wbf;
    __bf16* wb_l   = wbf + 16384;
    __bf16* wb_src = wbf + 2 * 16384;
    __bf16* wb_dst = wbf + 3 * 16384;
    __bf16* wb_out = wbf + 4 * 16384;

    float invN = 1.0f / (float)N;
    float invE = 1.0f / (float)E;
    int gM = (N + 31) / 32;

    zero_f_kernel<<<(nStats + 255) / 256, 256, 0, stream>>>(ws, nStats);
    f32_to_bf16_kernel<<<64, 256, 0, stream>>>(W_in,  wb_in,  16384);
    f32_to_bf16_kernel<<<64, 256, 0, stream>>>(Wl,    wb_l,   16384);
    f32_to_bf16_kernel<<<64, 256, 0, stream>>>(Wsrc,  wb_src, 16384);
    f32_to_bf16_kernel<<<64, 256, 0, stream>>>(Wdst,  wb_dst, 16384);
    f32_to_bf16_kernel<<<64, 256, 0, stream>>>(W_out, wb_out, 16384);

    // h1 = x @ W_in^T  (raw, pre-BN)
    gemm_f32A_wmma_kernel<false><<<gM, 256, 0, stream>>>(
        x, wb_in, nullptr, nullptr, nullptr, nullptr, nullptr, 0.0f, h1, N);
    colstats128_kernel<<<512, 128, 0, stream>>>(h1, N, S1s, S1q);

    // h = relu(bn1(h1)) materialized once in bf16; three bf16 GEMMs share it
    bnrelu_bf16_kernel<<<1024, 256, 0, stream>>>(
        h1, S1s, S1q, bn1_g, bn1_b, invN, hbf, N * 128);
    gemm_bf16A_wmma_kernel<<<gM, 256, 0, stream>>>(hbf, wb_l,   bl,   xlin,  N);
    gemm_bf16A_wmma_kernel<<<gM, 256, 0, stream>>>(hbf, wb_src, bsrc, asrcB, N);
    gemm_bf16A_wmma_kernel<<<gM, 256, 0, stream>>>(hbf, wb_dst, bdst, adstB, N);

    // edge pipeline
    edge_bnp_stats_kernel<<<1024, 256, 0, stream>>>(
        srcArr, dstArr, pos, Wp1, bp1, E, Ps, Pq);
    edge_bna1_stats_kernel<<<(N + NPB2 - 1) / NPB2, 128, 0, stream>>>(
        srcArr, pos, asrcB, adstB, Wp1, bp1, Wp2, bp2,
        Ps, Pq, bnp_g, bnp_b, invE, N, Kdeg, A1s, A1q);
    edge_t2_kernel<<<(N + NPB3 - 1) / NPB3, 128, 0, stream>>>(
        srcArr, pos, asrcB, adstB, Wp1, bp1, Wp2, bp2,
        Ps, Pq, bnp_g, bnp_b, A1s, A1q, bna1_g, bna1_b,
        Wa1, ba1, invE, N, Kdeg, t2, A2s, A2q);
    edge_attn_agg_kernel<<<(N + NPB4 - 1) / NPB4, 128, 0, stream>>>(
        srcArr, pos, xlin, t2, Wp1, bp1, Wp2, bp2,
        Ps, Pq, bnp_g, bnp_b, A2s, A2q, bna2_g, bna2_b,
        Wa2, ba2, invE, N, Kdeg, agg);

    // tail: bn2+relu fused into W_out GEMM A-load, then bn3 + residual + relu
    colstats128_kernel<<<512, 128, 0, stream>>>(agg, N, S2s, S2q);
    gemm_f32A_wmma_kernel<true><<<gM, 256, 0, stream>>>(
        agg, wb_out, nullptr, S2s, S2q, bn2_g, bn2_b, invN, h3, N);
    colstats128_kernel<<<512, 128, 0, stream>>>(h3, N, S3s, S3q);
    final_bn_res_kernel<<<1024, 256, 0, stream>>>(
        h3, x, S3s, S3q, bn3_g, bn3_b, invN, (float*)d_out, N * 128);
}
